// LSSModel_7610682049041
// MI455X (gfx1250) — matrix operand order repaired and checked by reading, not compiled, not run on hardware
//
#include <hip/hip_runtime.h>
#include <hip/hip_bf16.h>

typedef __attribute__((ext_vector_type(16))) __bf16 v16bf;
typedef __attribute__((ext_vector_type(8)))  __bf16 v8bf;
typedef __attribute__((ext_vector_type(8)))  float  v8f;

#define BB    4
#define NN    6
#define BN_   24
#define CC    64
#define DD    41
#define IMGH  384
#define IMGW  1056
#define FH_   12
#define FW_   33
#define NPIX  396          // FH_*FW_
#define NO    105          // DD + CC
#define NX_   200
#define NY_   200
#define PAD_H 202
#define PAD_W 216
#define NPTS  (BN_*DD*NPIX)           // 389664
#define NCLS_ 4
#define GRID_E ((long long)BB*NY_*NX_*CC)        // 10,240,000 f32
#define PAD_E  ((long long)BB*PAD_H*PAD_W*CC)    // 11,169,792 bf16
#define PW_E   (9*2*4*32*16)                     // 36,864 bf16

// ---------------------------------------------------------------- zero
__global__ void zero_kernel(unsigned int* p, long long n) {
  long long i = (long long)blockIdx.x * blockDim.x + threadIdx.x;
  long long stride = (long long)gridDim.x * blockDim.x;
  for (; i < n; i += stride) p[i] = 0u;
}

// ---------------------------------------------------------------- 32x32 avg pool
__global__ void pool_kernel(const float* __restrict__ imgs, float* __restrict__ pooled) {
  int ob = blockIdx.x;                 // ((bn*3+c)*FH + h)*FW + w
  int w = ob % FW_; int t = ob / FW_;
  int h = t % FH_;  t /= FH_;
  int c = t % 3;    int bn = t / 3;
  const float* src = imgs + (((long long)(bn*3 + c)*IMGH + h*32)*IMGW + w*32);
  float s = 0.f;
  for (int e = threadIdx.x; e < 1024; e += 256) {
    int r = e >> 5, col = e & 31;
    s += src[(long long)r*IMGW + col];
  }
  __shared__ float red[256];
  red[threadIdx.x] = s;
  __syncthreads();
  for (int k = 128; k > 0; k >>= 1) {
    if (threadIdx.x < k) red[threadIdx.x] += red[threadIdx.x + k];
    __syncthreads();
  }
  if (threadIdx.x == 0) pooled[ob] = red[0] * (1.f/1024.f);
}

// ---------------------------------------------------------------- cam MLP + softmax
__global__ void camfeat_kernel(const float* __restrict__ pooled,
                               const float* __restrict__ cam_w,
                               const float* __restrict__ cam_b,
                               float* __restrict__ depth,   // (bn, D, pix)
                               float* __restrict__ ctx) {   // (bn, pix, C) channel-last
  int q = blockIdx.x*blockDim.x + threadIdx.x;
  if (q >= BN_*NPIX) return;
  int bn = q / NPIX, pix = q % NPIX;
  float p0 = pooled[(bn*3+0)*NPIX + pix];
  float p1 = pooled[(bn*3+1)*NPIX + pix];
  float p2 = pooled[(bn*3+2)*NPIX + pix];
  float f[DD];
  #pragma unroll
  for (int o = 0; o < DD; ++o)
    f[o] = cam_b[o] + cam_w[o*3]*p0 + cam_w[o*3+1]*p1 + cam_w[o*3+2]*p2;
  float m = f[0];
  #pragma unroll
  for (int o = 1; o < DD; ++o) m = fmaxf(m, f[o]);
  float sum = 0.f;
  #pragma unroll
  for (int o = 0; o < DD; ++o) { f[o] = __expf(f[o]-m); sum += f[o]; }
  float inv = 1.f / sum;
  #pragma unroll
  for (int o = 0; o < DD; ++o) depth[(bn*DD+o)*NPIX + pix] = f[o]*inv;
  #pragma unroll
  for (int c = 0; c < CC; ++c) {
    int o = DD + c;
    float v = cam_b[o] + cam_w[o*3]*p0 + cam_w[o*3+1]*p1 + cam_w[o*3+2]*p2;
    ctx[(long long)q*CC + c] = v;
  }
}

// ---------------------------------------------------------------- frustum geometry
__global__ void geom_kernel(const float* __restrict__ rots,
                            const float* __restrict__ trans,
                            const float* __restrict__ intr,
                            int* __restrict__ vox) {
  int p = blockIdx.x*blockDim.x + threadIdx.x;
  if (p >= NPTS) return;
  int bn  = p / (DD*NPIX);
  int rem = p % (DD*NPIX);
  int dep = rem / NPIX;
  int pix = rem % NPIX;
  int h = pix / FW_, w = pix % FW_;
  float xs = (float)w * ((float)(IMGW-1) / (float)(FW_-1));
  float ys = (float)h * ((float)(IMGH-1) / (float)(FH_-1));
  float d  = 4.f + (float)dep;
  const float* R  = rots  + bn*9;
  const float* Tt = trans + bn*3;
  const float* K  = intr  + bn*9;
  float fx = K[0], cx = K[2], fy = K[4], cy = K[5];
  float px = (xs - cx) * d / fx;
  float py = (ys - cy) * d / fy;
  float wx = R[0]*px + R[1]*py + R[2]*d + Tt[0];
  float wy = R[3]*px + R[4]*py + R[5]*d + Tt[1];
  float wz = R[6]*px + R[7]*py + R[8]*d + Tt[2];
  int ix = (int)floorf((wx + 50.f) * 2.f);
  int iy = (int)floorf((wy + 50.f) * 2.f);
  int iz = (int)floorf((wz + 10.f) * (1.f/20.f));
  bool valid = (ix >= 0) && (ix < NX_) && (iy >= 0) && (iy < NY_) && (iz == 0);
  int b = bn / NN;
  vox[p] = valid ? ((b*NY_ + iy)*NX_ + ix) : -1;
}

// ---------------------------------------------------------------- splat (atomic scatter)
__global__ void scatter_kernel(const int* __restrict__ vox,
                               const float* __restrict__ depth,
                               const float* __restrict__ ctx,
                               float* __restrict__ grid) {
  long long gid = (long long)blockIdx.x*blockDim.x + threadIdx.x;
  if (gid >= (long long)NPTS*CC) return;
  int c = (int)(gid & 63);
  int p = (int)(gid >> 6);
  int v = vox[p];
  if (v < 0) return;
  int bn  = p / (DD*NPIX);
  int rem = p % (DD*NPIX);
  int dep = rem / NPIX;
  int pix = rem % NPIX;
  float dval = depth[(bn*DD + dep)*NPIX + pix];
  float cval = ctx[((long long)bn*NPIX + pix)*CC + c];
  unsafeAtomicAdd(&grid[(long long)v*CC + c], cval * dval);
}

// ---------------------------------------------------------------- f32 -> padded bf16
__global__ void convert_kernel(const float* __restrict__ grid, __bf16* __restrict__ pad) {
  long long e = (long long)blockIdx.x*blockDim.x + threadIdx.x;
  if (e >= PAD_E) return;
  int c = (int)(e & 63); long long t = e >> 6;
  int x = (int)(t % PAD_W); t /= PAD_W;
  int y = (int)(t % PAD_H); int b = (int)(t / PAD_H);
  float v = 0.f;
  if (y >= 1 && y <= NY_ && x >= 1 && x <= NX_)
    v = grid[(((long long)b*NY_ + (y-1))*NX_ + (x-1))*CC + c];
  pad[e] = (__bf16)v;
}

// ---------------------------------------------------------------- pack conv weights into WMMA B-fragment layout
__global__ void prepw_kernel(const float* __restrict__ comp_w, __bf16* __restrict__ pw) {
  int i = blockIdx.x*blockDim.x + threadIdx.x;
  if (i >= PW_E) return;
  int e = i & 15;  int t = i >> 4;
  int lane = t & 31; t >>= 5;
  int nt = t & 3;  t >>= 2;
  int s = t & 1;   int tap = t >> 1;
  int ky = tap / 3, kx = tap % 3;
  int n  = lane & 15;
  int kk = ((lane < 16) ? 0 : 8) + ((e < 8) ? e : 8 + e);   // lanes 0-15 K=0..7,16..23; lanes 16-31 K=8..15,24..31
  int o  = nt*16 + n;
  int ci = s*32 + kk;
  pw[i] = (__bf16)comp_w[((o*CC + ci)*3 + ky)*3 + kx];
}

// ---------------------------------------------------------------- 3x3 conv as implicit GEMM on WMMA + fused BN/ReLU
__global__ void __launch_bounds__(256) conv_wmma_kernel(
    const __bf16* __restrict__ pad, const __bf16* __restrict__ pw,
    const float* __restrict__ comp_b, const float* __restrict__ gamma,
    const float* __restrict__ beta,  const float* __restrict__ mean,
    const float* __restrict__ var,   float* __restrict__ zbuf) {
  __shared__ __attribute__((aligned(32))) __bf16 ldsW[PW_E];   // 73,728 B of 320 KB WGP LDS

  // stage weights to LDS via the gfx1250 async engine (no VGPR round-trip):
  //   global_load_async_to_lds_b128  (tracked by ASYNCcnt, ISA 08 §4)
  {
    unsigned ldsBase = (unsigned)(uintptr_t)ldsW;
    unsigned long long gBase = (unsigned long long)(uintptr_t)pw;
    // (PW_E*2)/16 = 4608 b128 transfers, 256 threads -> 18 each
    for (int i = threadIdx.x; i < (PW_E*2)/16; i += 256) {
      unsigned ldst = ldsBase + (unsigned)(i*16);
      unsigned long long ga = gBase + (unsigned long long)(i*16);
      asm volatile("global_load_async_to_lds_b128 %0, %1, off"
                   :: "v"(ldst), "v"(ga) : "memory");
    }
    asm volatile("s_wait_asynccnt 0x0" ::: "memory");
  }
  __syncthreads();

  int blk = blockIdx.x;            // B*NY_*2 blocks
  int xb = blk & 1; int t = blk >> 1;
  int y = t % NY_;  int b = t / NY_;
  int wid  = threadIdx.x >> 5;
  int lane = threadIdx.x & 31;
  int x0   = xb*128 + wid*16;      // 16-pixel M-tile per wave
  int mrow = lane & 15;
  int hi   = lane >> 4;            // A K-half select (K 0..7/16..23 vs 8..15/24..31)

  v8f acc[4] = {};                 // 4 N-tiles of 16 -> full 64 output channels

  #pragma unroll
  for (int tap = 0; tap < 9; ++tap) {
    int ky = tap / 3, kx = tap % 3;
    int pxl = x0 + mrow + kx;
    if (pxl > PAD_W - 1) pxl = PAD_W - 1;              // keep OOB waves in-allocation
    const __bf16* abase = pad + (((long long)b*PAD_H + (y + ky))*PAD_W + pxl)*CC;
    __builtin_prefetch(abase + (long long)PAD_W*CC, 0, 1);  // next row -> global_prefetch_b8
    #pragma unroll
    for (int s = 0; s < 2; ++s) {
      const __bf16* ap = abase + s*32 + hi*8;
      v8bf lo = *(const v8bf*)(ap);
      v8bf hh = *(const v8bf*)(ap + 16);
      v16bf A = __builtin_shufflevector(lo, hh,
                  0, 1, 2, 3, 4, 5, 6, 7, 8, 9, 10, 11, 12, 13, 14, 15);
      int wbase = ((tap*2 + s)*4)*512 + lane*16;
      #pragma unroll
      for (int nt = 0; nt < 4; ++nt) {
        v16bf Bf = *(const v16bf*)(ldsW + wbase + nt*512);
        acc[nt] = __builtin_amdgcn_wmma_f32_16x16x32_bf16(
            false, A, false, Bf, (short)0, acc[nt], false, false);
      }
    }
  }

  // epilogue: bias + batchnorm + relu, store channel-last z
  #pragma unroll
  for (int nt = 0; nt < 4; ++nt) {
    int o = nt*16 + mrow;                       // C/D layout: N = lane&15
    float sc  = gamma[o] * rsqrtf(var[o] + 1e-5f);
    float bsh = (comp_b[o] - mean[o]) * sc + beta[o];
    #pragma unroll
    for (int r = 0; r < 8; ++r) {
      int m = hi*8 + r;                         // C/D layout: M = (lane>=16 ? 8 : 0) + vgpr
      int pxo = x0 + m;
      if (pxo < NX_) {
        float zv = fmaxf(acc[nt][r]*sc + bsh, 0.f);
        zbuf[(((long long)b*NY_ + y)*NX_ + pxo)*CC + o] = zv;
      }
    }
  }
}

// ---------------------------------------------------------------- 4x64 decoder (coalesced + LDS reduce)
__global__ void __launch_bounds__(256) decoder_kernel(
    const float* __restrict__ zbuf, const float* __restrict__ dec_w,
    const float* __restrict__ dec_b, float* __restrict__ out) {
  int tid = threadIdx.x;
  int lp  = tid >> 6;                 // local pixel 0..3
  int c   = tid & 63;                 // channel
  int pixel = blockIdx.x*4 + lp;      // global pixel over B*NY*NX
  bool inb = pixel < BB*NY_*NX_;
  float zv = inb ? zbuf[(long long)pixel*CC + c] : 0.f;   // coalesced b32 per 64-group
  __shared__ float red[4][NCLS_][CC];
  #pragma unroll
  for (int cls = 0; cls < NCLS_; ++cls) red[lp][cls][c] = zv * dec_w[cls*CC + c];
  __syncthreads();
  for (int k = 32; k >= 1; k >>= 1) {
    if (c < k) {
      #pragma unroll
      for (int cls = 0; cls < NCLS_; ++cls) red[lp][cls][c] += red[lp][cls][c + k];
    }
    __syncthreads();
  }
  if (c < NCLS_ && inb) {
    int b  = pixel / (NY_*NX_);
    int yx = pixel % (NY_*NX_);
    out[((long long)b*NCLS_ + c)*(NY_*NX_) + yx] = red[lp][c][0] + dec_b[c];
  }
}

// ---------------------------------------------------------------- host
extern "C" void kernel_launch(void* const* d_in, const int* in_sizes, int n_in,
                              void* d_out, int out_size, void* d_ws, size_t ws_size,
                              hipStream_t stream) {
  const float* imgs   = (const float*)d_in[0];
  const float* rots   = (const float*)d_in[1];
  const float* trans  = (const float*)d_in[2];
  const float* intr   = (const float*)d_in[3];
  const float* cam_w  = (const float*)d_in[4];
  const float* cam_b  = (const float*)d_in[5];
  const float* comp_w = (const float*)d_in[6];
  const float* comp_b = (const float*)d_in[7];
  const float* gamma  = (const float*)d_in[8];
  const float* beta   = (const float*)d_in[9];
  const float* mean   = (const float*)d_in[10];
  const float* var    = (const float*)d_in[11];
  const float* dec_w  = (const float*)d_in[12];
  const float* dec_b  = (const float*)d_in[13];
  float* out = (float*)d_out;

  char* ws = (char*)d_ws;
  auto al = [](size_t x){ return (x + 255) & ~(size_t)255; };
  size_t off = 0;
  float*  grid   = (float*)(ws + off);  off += al(sizeof(float)*GRID_E);   // reused as zbuf
  __bf16* pad    = (__bf16*)(ws + off); off += al(2ull*PAD_E);
  __bf16* pw     = (__bf16*)(ws + off); off += al(2ull*PW_E);
  float*  pooled = (float*)(ws + off);  off += al(sizeof(float)*BN_*3*NPIX);
  float*  depth  = (float*)(ws + off);  off += al(sizeof(float)*(size_t)BN_*DD*NPIX);
  float*  ctx    = (float*)(ws + off);  off += al(sizeof(float)*(size_t)BN_*NPIX*CC);
  int*    vox    = (int*)(ws + off);    off += al(sizeof(int)*(size_t)NPTS);

  zero_kernel<<<2048, 256, 0, stream>>>((unsigned int*)grid, GRID_E);

  pool_kernel<<<BN_*3*FH_*FW_, 256, 0, stream>>>(imgs, pooled);

  camfeat_kernel<<<(BN_*NPIX + 255)/256, 256, 0, stream>>>(pooled, cam_w, cam_b, depth, ctx);

  geom_kernel<<<(NPTS + 255)/256, 256, 0, stream>>>(rots, trans, intr, vox);

  scatter_kernel<<<(int)(((long long)NPTS*CC + 255)/256), 256, 0, stream>>>(vox, depth, ctx, grid);

  prepw_kernel<<<(PW_E + 255)/256, 256, 0, stream>>>(comp_w, pw);

  convert_kernel<<<(int)((PAD_E + 255)/256), 256, 0, stream>>>(grid, pad);

  // conv writes z into the (now converted) grid buffer
  conv_wmma_kernel<<<BB*NY_*2, 256, 0, stream>>>(pad, pw, comp_b, gamma, beta, mean, var, grid);

  decoder_kernel<<<(BB*NY_*NX_ + 3)/4, 256, 0, stream>>>(grid, dec_w, dec_b, out);
}
